// RnnEncoder_63677185131185
// MI455X (gfx1250) — compile-verified, matrix operand order checked by reading, not run on hardware
//
#include <hip/hip_runtime.h>

#define B_ 64
#define T_ 2048
#define E_ 256
#define H_ 512

typedef __bf16 v16bf __attribute__((ext_vector_type(16)));
typedef __bf16 v8bf  __attribute__((ext_vector_type(8)));
typedef float  v8f   __attribute__((ext_vector_type(8)));
typedef float  v4f   __attribute__((ext_vector_type(4)));

#if __has_builtin(__builtin_amdgcn_tensor_load_to_lds)
#define HAVE_TDM 1
typedef unsigned int u32x4 __attribute__((ext_vector_type(4)));
typedef int          i32x4 __attribute__((ext_vector_type(4)));
typedef int          i32x8 __attribute__((ext_vector_type(8)));
#endif

__device__ __forceinline__ float tanh_fast(float x) {
    // tanh(x) = 1 - 2/(exp2(2x*log2e)+1); v_exp_f32 + v_rcp_f32 trans units
    float e = __builtin_amdgcn_exp2f(x * 2.885390081777927f);
    return 1.0f - 2.0f * __builtin_amdgcn_rcpf(e + 1.0f);
}

// Load a 16x32 fp32 tile chunk in WMMA 16-bit A/B lane layout, convert to bf16.
__device__ __forceinline__ v16bf load_tile_bf16(const float* __restrict__ base,
                                                int ldk, int row, int k0) {
    const float* p = base + (size_t)row * ldk + k0;
    v4f a0 = *(const v4f*)p,        a1 = *(const v4f*)(p + 4);
    v4f a2 = *(const v4f*)(p + 16), a3 = *(const v4f*)(p + 20);
    v16bf v;
#pragma unroll
    for (int j = 0; j < 4; ++j) {
        v[j]      = (__bf16)a0[j]; v[4 + j]  = (__bf16)a1[j];
        v[8 + j]  = (__bf16)a2[j]; v[12 + j] = (__bf16)a3[j];
    }
    return v;
}

__device__ __forceinline__ void gridBarrier(unsigned* arrive, unsigned* release,
                                            unsigned nwg, unsigned& phase) {
    __syncthreads();
    if (threadIdx.x == 0) {
        __builtin_amdgcn_fence(__ATOMIC_RELEASE, "agent");
        unsigned prev = __hip_atomic_fetch_add(arrive, 1u, __ATOMIC_RELAXED,
                                               __HIP_MEMORY_SCOPE_AGENT);
        if (prev == phase * nwg + (nwg - 1u)) {
            __hip_atomic_store(release, phase + 1u, __ATOMIC_RELAXED,
                               __HIP_MEMORY_SCOPE_AGENT);
        }
        while (__hip_atomic_load(release, __ATOMIC_RELAXED,
                                 __HIP_MEMORY_SCOPE_AGENT) <= phase) {
            __builtin_amdgcn_s_sleep(1);
        }
        __builtin_amdgcn_fence(__ATOMIC_ACQUIRE, "agent");
    }
    ++phase;
    __syncthreads();
}

// ---------------- init --------------------------------------------------------
__global__ void init_kernel(__bf16* hbuf, float* logit, unsigned* bar) {
    int i = blockIdx.x * blockDim.x + threadIdx.x;
    const int n = 2 * B_ * H_;
    for (int j = i; j < n; j += gridDim.x * blockDim.x) hbuf[j] = (__bf16)0.0f;
    if (i < 2 * B_) logit[i] = 0.0f;
    if (i < 2) bar[i] = 0u;
}

// ---------------- phase 1: xin = x @ W_ih^T + b_ih + b_hh  (bf16 out) --------
// Register-blocked: each wave owns a 32x32 output (2x2 WMMA tiles), so each
// A/B tile load feeds 2 WMMAs. Block = 128(M) x 64(N), 8 waves.
__global__ void __launch_bounds__(256)
xin_gemm(const float* __restrict__ x, const float* __restrict__ Wih,
         const float* __restrict__ bih, const float* __restrict__ bhh,
         __bf16* __restrict__ xin) {
    const int lane = threadIdx.x & 31;
    const int wave = threadIdx.x >> 5;
    const int bn = blockIdx.x & 7;          // 512/64 = 8 N-blocks
    const int bm = blockIdx.x >> 3;         // 131072/128 = 1024 M-blocks
    const int m_base = bm * 128 + (wave >> 1) * 32;
    const int n_base = bn * 64 + (wave & 1) * 32;
    const int lrow = lane & 15;
    const int koff = (lane < 16) ? 0 : 8;

    v8f acc00 = {}, acc01 = {}, acc10 = {}, acc11 = {};
#pragma unroll
    for (int kk = 0; kk < E_ / 32; ++kk) {
        const int k0 = kk * 32 + koff;
        v16bf va0 = load_tile_bf16(x,   E_, m_base + lrow,      k0);
        v16bf va1 = load_tile_bf16(x,   E_, m_base + 16 + lrow, k0);
        v16bf vb0 = load_tile_bf16(Wih, E_, n_base + lrow,      k0);
        v16bf vb1 = load_tile_bf16(Wih, E_, n_base + 16 + lrow, k0);
        acc00 = __builtin_amdgcn_wmma_f32_16x16x32_bf16(false, va0, false, vb0,
                                                        (short)0, acc00, false, false);
        acc01 = __builtin_amdgcn_wmma_f32_16x16x32_bf16(false, va0, false, vb1,
                                                        (short)0, acc01, false, false);
        acc10 = __builtin_amdgcn_wmma_f32_16x16x32_bf16(false, va1, false, vb0,
                                                        (short)0, acc10, false, false);
        acc11 = __builtin_amdgcn_wmma_f32_16x16x32_bf16(false, va1, false, vb1,
                                                        (short)0, acc11, false, false);
    }
    const int mhalf = (lane < 16) ? 0 : 8;
#pragma unroll
    for (int nj = 0; nj < 2; ++nj) {
        const int col = n_base + nj * 16 + lrow;
        const float bias = bih[col] + bhh[col];
#pragma unroll
        for (int mi = 0; mi < 2; ++mi) {
            const v8f& a = (mi == 0) ? (nj == 0 ? acc00 : acc01)
                                     : (nj == 0 ? acc10 : acc11);
            const int mb = m_base + mi * 16 + mhalf;
#pragma unroll
            for (int i = 0; i < 8; ++i)
                xin[(size_t)(mb + i) * H_ + col] = (__bf16)(a[i] + bias);
        }
    }
}

// ---------------- phase 2: persistent sequential scan ------------------------
// 32 WGs; WG g owns H columns [16g,16g+16). W_hh slice resident in LDS.
// Waves 0-3: matmul+tanh for step t (two independent WMMA chains for ILP).
// Wave 7: TDM-prefetch next step's xin tile into double-buffered LDS.
// Waves 4-7: fc partials for step t-1; WG0/wave4 counts signs for t-2.
__global__ void __launch_bounds__(256)
rnn_scan(const __bf16* __restrict__ xin, __bf16* hbuf, float* logit,
         unsigned* bar, const float* __restrict__ Whh,
         const int* __restrict__ lengths, const float* __restrict__ wfc,
         const float* __restrict__ bfcp, int* __restrict__ outs,
         float* __restrict__ hN) {
    __shared__ __bf16 Wl[16 * H_];
    __shared__ __bf16 xin_lds[2][B_][16];   // double-buffered step tile
    __shared__ float  wfc_l[16];
    __shared__ int    len_l[B_];

    const int tid  = threadIdx.x;
    const int lane = tid & 31;
    const int wave = tid >> 5;
    const int n0   = blockIdx.x * 16;

    for (int i = tid; i < 16 * H_; i += 256) {
        int r = i >> 9, k = i & (H_ - 1);
        Wl[i] = (__bf16)Whh[(size_t)(n0 + r) * H_ + k];
    }
    if (tid < 16)  wfc_l[tid] = wfc[n0 + tid];
    if (tid < B_)  len_l[tid] = lengths[tid];
    const float bfc = bfcp[0];

#ifdef HAVE_TDM
    // Wave 7 drives the Tensor Data Mover: gather 64 rows x 16 bf16 (stride T*H)
    // of xin for a given step into LDS.
    if (wave == 7 && lane == 0) {
        unsigned long long ga =
            (unsigned long long)(const void*)(xin + (size_t)0 * H_ + n0);
        unsigned lds_off =
            (unsigned)(unsigned long long)(const void*)&xin_lds[0][0][0];
        u32x4 g0;
        g0[0] = 1u;                                       // count=1
        g0[1] = lds_off;                                  // lds_addr
        g0[2] = (unsigned)(ga & 0xFFFFFFFFu);             // global_addr[31:0]
        g0[3] = (unsigned)((ga >> 32) & 0x1FFFFFFu) | (2u << 30); // addr hi, type=2
        i32x8 g1;
        g1[0] = (1 << 16);          // data_size = 1 -> 2 bytes
        g1[1] = (16 << 16);         // tensor_dim0 = 16
        g1[2] = (B_ << 16);         // tensor_dim1 = 64
        g1[3] = (16 << 16);         // tile_dim0 = 16
        g1[4] = B_;                 // tile_dim1 = 64, tile_dim2 = 0
        g1[5] = T_ * H_;            // tensor_dim0_stride = T*H elements
        g1[6] = 0; g1[7] = 0;
        i32x4 gz = {0, 0, 0, 0};
#if __clang_major__ >= 23
        i32x8 gz8 = {0, 0, 0, 0, 0, 0, 0, 0};
        __builtin_amdgcn_tensor_load_to_lds(g0, g1, gz, gz, gz8, 0);
#else
        __builtin_amdgcn_tensor_load_to_lds(g0, g1, gz, gz, 0);
#endif
        __builtin_amdgcn_s_wait_tensorcnt(0);
    }
#else
    if (wave == 7) {  // fallback: plain-load the t=0 tile
        for (int i = lane; i < B_ * 16; i += 32)
            xin_lds[0][i >> 4][i & 15] = xin[((size_t)(i >> 4) * T_) * H_ + n0 + (i & 15)];
    }
#endif
    __syncthreads();

    unsigned phase = 0;
    int cnt0 = 0, cnt1 = 0;

    for (int t = 0; t < T_ + 2; ++t) {
        if (wave < 4) {
            if (t < T_) {
                const int m0   = wave * 16;
                const __bf16* hprev = hbuf + (size_t)((t + 1) & 1) * B_ * H_;
                const int rowA = m0 + (lane & 15);
                const int koff = (lane < 16) ? 0 : 8;
                const int r    = lane & 15;
                v8f acc0 = {}, acc1 = {};   // two chains -> WMMA ILP
#pragma unroll 4
                for (int kk = 0; kk < H_ / 32; kk += 2) {
                    const int kA = kk * 32 + koff;
                    const int kB = kA + 32;
                    v8bf a0l = *(const v8bf*)(hprev + (size_t)rowA * H_ + kA);
                    v8bf a0h = *(const v8bf*)(hprev + (size_t)rowA * H_ + kA + 16);
                    v8bf b0l = *(const v8bf*)(&Wl[r * H_ + kA]);
                    v8bf b0h = *(const v8bf*)(&Wl[r * H_ + kA + 16]);
                    v8bf a1l = *(const v8bf*)(hprev + (size_t)rowA * H_ + kB);
                    v8bf a1h = *(const v8bf*)(hprev + (size_t)rowA * H_ + kB + 16);
                    v8bf b1l = *(const v8bf*)(&Wl[r * H_ + kB]);
                    v8bf b1h = *(const v8bf*)(&Wl[r * H_ + kB + 16]);
                    v16bf va0, vb0, va1, vb1;
#pragma unroll
                    for (int j = 0; j < 8; ++j) {
                        va0[j] = a0l[j]; va0[8 + j] = a0h[j];
                        vb0[j] = b0l[j]; vb0[8 + j] = b0h[j];
                        va1[j] = a1l[j]; va1[8 + j] = a1h[j];
                        vb1[j] = b1l[j]; vb1[8 + j] = b1h[j];
                    }
                    acc0 = __builtin_amdgcn_wmma_f32_16x16x32_bf16(
                        false, va0, false, vb0, (short)0, acc0, false, false);
                    acc1 = __builtin_amdgcn_wmma_f32_16x16x32_bf16(
                        false, va1, false, vb1, (short)0, acc1, false, false);
                }
                const int col = n0 + (lane & 15);
                const int mb  = m0 + ((lane < 16) ? 0 : 8);
                __bf16* hcur  = hbuf + (size_t)(t & 1) * B_ * H_;
#pragma unroll
                for (int i = 0; i < 8; ++i) {
                    const int b = mb + i;
                    float xv = (float)xin_lds[t & 1][b][lane & 15];
                    float hv = tanh_fast(acc0[i] + acc1[i] + xv);
                    hcur[(size_t)b * H_ + col] = (__bf16)hv;
                    if (t == len_l[b] - 1) hN[(size_t)b * H_ + col] = hv;
                }
            }
        } else {
            // TDM prefetch of next step's xin tile (wave 7)
            if (wave == 7 && t + 1 < T_) {
#ifdef HAVE_TDM
                if (lane == 0) {
                    unsigned long long ga = (unsigned long long)(const void*)
                        (xin + (size_t)(t + 1) * H_ + n0);
                    unsigned lds_off = (unsigned)(unsigned long long)(const void*)
                        &xin_lds[(t + 1) & 1][0][0];
                    u32x4 g0;
                    g0[0] = 1u;
                    g0[1] = lds_off;
                    g0[2] = (unsigned)(ga & 0xFFFFFFFFu);
                    g0[3] = (unsigned)((ga >> 32) & 0x1FFFFFFu) | (2u << 30);
                    i32x8 g1;
                    g1[0] = (1 << 16);
                    g1[1] = (16 << 16);
                    g1[2] = (B_ << 16);
                    g1[3] = (16 << 16);
                    g1[4] = B_;
                    g1[5] = T_ * H_;
                    g1[6] = 0; g1[7] = 0;
                    i32x4 gz = {0, 0, 0, 0};
#if __clang_major__ >= 23
                    i32x8 gz8 = {0, 0, 0, 0, 0, 0, 0, 0};
                    __builtin_amdgcn_tensor_load_to_lds(g0, g1, gz, gz, gz8, 0);
#else
                    __builtin_amdgcn_tensor_load_to_lds(g0, g1, gz, gz, 0);
#endif
                }
#else
                for (int i = lane; i < B_ * 16; i += 32)
                    xin_lds[(t + 1) & 1][i >> 4][i & 15] =
                        xin[((size_t)(i >> 4) * T_ + (t + 1)) * H_ + n0 + (i & 15)];
#endif
            }
            if (blockIdx.x == 0 && wave == 4 && t >= 2) {
                const int s = t - 2;
                float* lg = logit + (size_t)(s & 1) * B_;
                float v0 = lg[lane] + bfc;
                float v1 = lg[lane + 32] + bfc;
                if (v0 > 0.0f && s < len_l[lane])      cnt0++;
                if (v1 > 0.0f && s < len_l[lane + 32]) cnt1++;
                lg[lane] = 0.0f; lg[lane + 32] = 0.0f;
            }
            const int s = t - 1;
            if (s >= 0 && s < T_) {
                const int wf = wave - 4;
                const int b  = wf * 16 + (lane >> 1);
                const int c0 = (lane & 1) * 8;
                const __bf16* hs = hbuf + (size_t)(s & 1) * B_ * H_ +
                                   (size_t)b * H_ + n0 + c0;
                float p = 0.0f;
#pragma unroll
                for (int j = 0; j < 8; ++j) p += (float)hs[j] * wfc_l[c0 + j];
                p += __shfl_xor(p, 1);
                if ((lane & 1) == 0) {
                    __hip_atomic_fetch_add(&logit[(size_t)(s & 1) * B_ + b], p,
                                           __ATOMIC_RELAXED, __HIP_MEMORY_SCOPE_AGENT);
                }
            }
#ifdef HAVE_TDM
            if (wave == 7 && t + 1 < T_) __builtin_amdgcn_s_wait_tensorcnt(0);
#endif
        }
        gridBarrier(bar, bar + 1, gridDim.x, phase);
    }

    if (blockIdx.x == 0 && wave == 4) {
        outs[lane]      = cnt0;
        outs[lane + 32] = cnt1;
    }
}

extern "C" void kernel_launch(void* const* d_in, const int* in_sizes, int n_in,
                              void* d_out, int out_size, void* d_ws, size_t ws_size,
                              hipStream_t stream) {
    const float* x       = (const float*)d_in[0];
    const int*   lengths = (const int*)d_in[1];
    const float* Wih     = (const float*)d_in[2];
    const float* Whh     = (const float*)d_in[3];
    const float* bih     = (const float*)d_in[4];
    const float* bhh     = (const float*)d_in[5];
    const float* wfc     = (const float*)d_in[6];
    const float* bfc     = (const float*)d_in[7];

    char* w = (char*)d_ws;
    const size_t xinBytes = (size_t)B_ * T_ * H_ * sizeof(__bf16);   // 128 MB
    __bf16*   xin   = (__bf16*)w;
    __bf16*   hbuf  = (__bf16*)(w + xinBytes);
    float*    logit = (float*)(w + xinBytes + (size_t)2 * B_ * H_ * sizeof(__bf16));
    unsigned* bar   = (unsigned*)(w + xinBytes + (size_t)2 * B_ * H_ * sizeof(__bf16) +
                                  (size_t)2 * B_ * sizeof(float));

    int*   outs = (int*)d_out;
    float* hN   = (float*)d_out + B_;

    init_kernel<<<64, 256, 0, stream>>>(hbuf, logit, bar);

    // 1024 M-blocks x 8 N-blocks, 128x64 per block
    xin_gemm<<<1024 * 8, 256, 0, stream>>>(x, Wih, bih, bhh, xin);

    rnn_scan<<<H_ / 16, 256, 0, stream>>>(xin, hbuf, logit, bar, Whh, lengths,
                                          wfc, bfc, outs, hN);
}
